// GraphConv_13546326851765
// MI455X (gfx1250) — matrix operand order repaired and checked by reading, not compiled, not run on hardware
//
#include <hip/hip_runtime.h>
#include <hip/hip_bf16.h>

#define CDIM 64
#define NFAC 4
#define R2MAX 47

typedef __attribute__((ext_vector_type(16))) _Float16 v16h;
typedef __attribute__((ext_vector_type(8)))  float    v8f;

// ---------------------------------------------------------------------------
// Per-hop small dense math (single block, 256 threads).
// Computes:
//   L1 = l@W1 + b1                         (NF x C)
//   Mt[f][k] = (W1 @ L1^T)[k][f]  (f16, padded to 16 rows)  -> for WMMA A
//   cvec[f]  = b1 . L1[f]
//   L2a = l@W2 + b2, Wt2 = weight@W2 + b2
//   srp = L2a @ Wt2^T  (NF x R)
//   A   = softmax(leaky(srp@Wwa + bwa))    (NF x R)
//   latent_new = A @ weight                (NF x C)  -> lat_mix (raw)
//   l_out = latent_new (stack1) or normalize(latent_new) (stack2)
//   lat_acc += l_out (stack2 only)
// ---------------------------------------------------------------------------
__global__ void setup_hop(
    const float* __restrict__ l_in,
    const float* __restrict__ W1, const float* __restrict__ b1,
    const float* __restrict__ W2, const float* __restrict__ b2,
    const float* __restrict__ Wwa, const float* __restrict__ bwa,
    const float* __restrict__ weight, int R,
    _Float16* __restrict__ Mt,
    float* __restrict__ cvec,
    float* __restrict__ lat_mix,
    float* __restrict__ l_out,
    float* __restrict__ lat_acc,
    int norm_l)
{
    __shared__ float sl [NFAC*CDIM];
    __shared__ float L1 [NFAC*CDIM];
    __shared__ float L2a[NFAC*CDIM];
    __shared__ float Wt2[R2MAX*CDIM];
    __shared__ float srp [NFAC*R2MAX];
    __shared__ float Arow[NFAC*R2MAX];
    __shared__ float lnew[NFAC*CDIM];
    __shared__ float rnorm[NFAC];

    const int t = threadIdx.x;            // 0..255
    const int f = t >> 6, c = t & 63;

    if (t < NFAC*CDIM) sl[t] = l_in[t];
    __syncthreads();

    {   // L1
        float acc = b1[c];
        for (int k = 0; k < CDIM; ++k) acc += sl[f*CDIM+k] * W1[k*CDIM+c];
        L1[f*CDIM+c] = acc;
    }
    {   // L2a
        float acc = b2[c];
        for (int k = 0; k < CDIM; ++k) acc += sl[f*CDIM+k] * W2[k*CDIM+c];
        L2a[f*CDIM+c] = acc;
    }
    for (int o = t; o < R*CDIM; o += 256) {   // Wt2
        int r = o >> 6, cc = o & 63;
        float acc = b2[cc];
        for (int k = 0; k < CDIM; ++k) acc += weight[r*CDIM+k] * W2[k*CDIM+cc];
        Wt2[o] = acc;
    }
    __syncthreads();

    {   // Mt[f][c] = sum_k W1[c][k]*L1[f][k]  (transposed M, f16)
        float acc = 0.f;
        for (int k = 0; k < CDIM; ++k) acc += W1[c*CDIM+k] * L1[f*CDIM+k];
        Mt[f*CDIM + c] = (_Float16)acc;
    }
    for (int o = t; o < (16-NFAC)*CDIM; o += 256) Mt[NFAC*CDIM + o] = (_Float16)0.f;
    if (t < NFAC) {
        float acc = 0.f;
        for (int k = 0; k < CDIM; ++k) acc += b1[k] * L1[t*CDIM+k];
        cvec[t] = acc;
    }
    for (int o = t; o < NFAC*R; o += 256) {   // srp = L2a @ Wt2^T
        int ff = o / R, r = o % R;
        float acc = 0.f;
        for (int k = 0; k < CDIM; ++k) acc += L2a[ff*CDIM+k] * Wt2[r*CDIM+k];
        srp[o] = acc;
    }
    __syncthreads();

    for (int o = t; o < NFAC*R; o += 256) {   // leaky(srp@Wwa + bwa)
        int ff = o / R, j = o % R;
        float acc = bwa[j];
        for (int r = 0; r < R; ++r) acc += srp[ff*R+r] * Wwa[r*R+j];
        Arow[o] = (acc >= 0.f) ? acc : 0.2f*acc;
    }
    __syncthreads();

    if (t < NFAC) {   // row softmax over R
        float mx = -1e30f;
        for (int j = 0; j < R; ++j) mx = fmaxf(mx, Arow[t*R+j]);
        float s = 0.f;
        for (int j = 0; j < R; ++j) { float e = __expf(Arow[t*R+j]-mx); Arow[t*R+j] = e; s += e; }
        float inv = 1.f/s;
        for (int j = 0; j < R; ++j) Arow[t*R+j] *= inv;
    }
    __syncthreads();

    {   // latent_new = A @ weight
        float acc = 0.f;
        for (int r = 0; r < R; ++r) acc += Arow[f*R+r] * weight[r*CDIM+c];
        lnew[f*CDIM+c] = acc;
        lat_mix[f*CDIM+c] = acc;
    }
    __syncthreads();

    if (norm_l) {
        if (t < NFAC) {
            float s = 0.f;
            for (int k = 0; k < CDIM; ++k) { float v = lnew[t*CDIM+k]; s += v*v; }
            rnorm[t] = 1.f / fmaxf(sqrtf(s), 1e-12f);
        }
        __syncthreads();
        float v = lnew[f*CDIM+c] * rnorm[f];
        l_out[f*CDIM+c] = v;
        if (lat_acc) lat_acc[f*CDIM+c] += v;
    } else {
        l_out[f*CDIM+c] = lnew[f*CDIM+c];
    }
}

// ---------------------------------------------------------------------------
// Edge scatter: one wave per edge, 2 channels per lane.
// neigh = e_cur[tail] * weight[idx]; atomic-add into ssum[head]; count heads.
// ---------------------------------------------------------------------------
__global__ void edge_agg(
    const float* __restrict__ e_cur,
    const float* __restrict__ weight,
    const int* __restrict__ head_arr,
    const int* __restrict__ tail_arr,
    const int* __restrict__ etype,
    const int* __restrict__ cate,
    float* __restrict__ ssum,
    float* __restrict__ cnt,
    int nE, int use_cate)
{
    long long gid = (long long)blockIdx.x * blockDim.x + threadIdx.x;
    long long e = gid >> 5;
    int lane = (int)(gid & 31);
    if (e >= nE) return;
    int h  = head_arr[e];
    int tl = tail_arr[e];
    int w  = use_cate ? (cate[tl] - 1) : (etype[e] - 1);
    int c = lane * 2;
    const float* er = e_cur + (long long)tl * CDIM;
    const float* wr = weight + (long long)w * CDIM;
    float v0 = er[c]   * wr[c];
    float v1 = er[c+1] * wr[c+1];
    float* sr = ssum + (long long)h * CDIM;
    atomicAdd(&sr[c],   v0);
    atomicAdd(&sr[c+1], v1);
    if (lane == 0) atomicAdd(&cnt[h], 1.0f);
}

// ---------------------------------------------------------------------------
// Interaction scatter: one wave per nnz.
// ---------------------------------------------------------------------------
__global__ void inter_agg(
    const float* __restrict__ e_cur,
    const float* __restrict__ vals,
    const int* __restrict__ rows,
    const int* __restrict__ cols,
    float* __restrict__ u_agg, int nnz)
{
    long long gid = (long long)blockIdx.x * blockDim.x + threadIdx.x;
    long long i = gid >> 5;
    int lane = (int)(gid & 31);
    if (i >= nnz) return;
    int r = rows[i], col = cols[i];
    float v = vals[i];
    int c = lane * 2;
    const float* er = e_cur + (long long)col * CDIM;
    float* ur = u_agg + (long long)r * CDIM;
    atomicAdd(&ur[c],   v * er[c]);
    atomicAdd(&ur[c+1], v * er[c+1]);
}

// ---------------------------------------------------------------------------
// Entity finalize: e_cur = normalize(ssum / max(cnt,1)); e_res += e_cur.
// One wave per entity.
// ---------------------------------------------------------------------------
__global__ void ent_finalize(
    const float* __restrict__ ssum,
    const float* __restrict__ cnt,
    float* __restrict__ e_cur,
    float* __restrict__ e_res, int n)
{
    long long gid = (long long)blockIdx.x * blockDim.x + threadIdx.x;
    long long ent = gid >> 5;
    int lane = (int)(gid & 31);
    if (ent >= n) return;
    float ic = 1.0f / fmaxf(cnt[ent], 1.0f);
    long long base = ent * CDIM;
    int c = lane * 2;
    float v0 = ssum[base+c]   * ic;
    float v1 = ssum[base+c+1] * ic;
    float s = v0*v0 + v1*v1;
    for (int m = 16; m > 0; m >>= 1) s += __shfl_xor(s, m, 32);
    float inv = 1.f / fmaxf(sqrtf(s), 1e-12f);
    v0 *= inv; v1 *= inv;
    e_cur[base+c]   = v0;
    e_cur[base+c+1] = v1;
    e_res[base+c]   += v0;
    e_res[base+c+1] += v1;
}

// ---------------------------------------------------------------------------
// User finalize (WMMA): per 16-user tile (one wave):
//   score_^T(16x16) = Mt(16x64,f16) x u_tile^T(64x16,f16) via 2x wmma k=32
//   lanes 0-15 own user n: acc[0..3] = score_[n][0..3]
//   -> +cvec, 4x4 Wua MLP, leaky, softmax -> sm[16][4] in LDS
//   phase 2: mix = sm @ lat_mix; u = normalize(u_agg*(1+mix)); u_res += u
// ---------------------------------------------------------------------------
__global__ void __launch_bounds__(32) user_finalize(
    float* __restrict__ u_cur,
    const float* __restrict__ u_agg,
    const _Float16* __restrict__ Mt,
    const float* __restrict__ cvec,
    const float* __restrict__ Wua,
    const float* __restrict__ bua,
    const float* __restrict__ lat_mix,
    float* __restrict__ u_res,
    int n_users)
{
    int u0 = blockIdx.x * 16;
    if (u0 >= n_users) return;
    int lane = threadIdx.x;
    int row  = lane & 15;
    int hi   = (lane >= 16) ? 1 : 0;
    const float* urow = u_cur + (long long)(u0 + row) * CDIM;

    v8f acc = {};
#pragma unroll
    for (int kb = 0; kb < 2; ++kb) {
        const int K0 = kb * 32;
        // A (16x32 f16): lanes0-15 K{0..7,16..23}, lanes16-31 K{8..15,24..31}
        const _Float16* mrow = Mt + row * CDIM + K0 + hi * 8;
        // B (32x16 f16): lanes0-15 K{0..7,16..23}, lanes16-31 K{8..15,24..31}
        const int kbB = K0 + hi * 8;
        v16h a, b;
#pragma unroll
        for (int i = 0; i < 8; ++i) {
            a[i]   = mrow[i];
            a[i+8] = mrow[16 + i];
        }
#pragma unroll
        for (int i = 0; i < 8; ++i) {
            b[i]   = (_Float16)urow[kbB + i];
            b[i+8] = (_Float16)urow[kbB + 16 + i];
        }
        acc = __builtin_amdgcn_wmma_f32_16x16x32_f16(
                false, a, false, b, (short)0, acc, false, false);
    }

    __shared__ float sm[16][NFAC];
    float s0 = acc[0] + cvec[0];
    float s1 = acc[1] + cvec[1];
    float s2 = acc[2] + cvec[2];
    float s3 = acc[3] + cvec[3];
    float tt[NFAC];
#pragma unroll
    for (int j = 0; j < NFAC; ++j) {
        float v = bua[j] + s0*Wua[0*NFAC+j] + s1*Wua[1*NFAC+j]
                         + s2*Wua[2*NFAC+j] + s3*Wua[3*NFAC+j];
        tt[j] = (v >= 0.f) ? v : 0.2f*v;
    }
    float mx = fmaxf(fmaxf(tt[0], tt[1]), fmaxf(tt[2], tt[3]));
    float e0 = __expf(tt[0]-mx), e1 = __expf(tt[1]-mx);
    float e2 = __expf(tt[2]-mx), e3 = __expf(tt[3]-mx);
    float inv = 1.f / (e0+e1+e2+e3);
    if (lane < 16) {
        sm[lane][0] = e0*inv; sm[lane][1] = e1*inv;
        sm[lane][2] = e2*inv; sm[lane][3] = e3*inv;
    }
    __syncthreads();

    int c = lane * 2;
    float lm0[NFAC], lm1[NFAC];
#pragma unroll
    for (int f = 0; f < NFAC; ++f) {
        lm0[f] = lat_mix[f*CDIM + c];
        lm1[f] = lat_mix[f*CDIM + c + 1];
    }
    for (int ul = 0; ul < 16; ++ul) {
        long long base = (long long)(u0 + ul) * CDIM;
        float a0 = u_agg[base+c], a1 = u_agg[base+c+1];
        float f0 = sm[ul][0], f1 = sm[ul][1], f2 = sm[ul][2], f3 = sm[ul][3];
        float m0 = f0*lm0[0] + f1*lm0[1] + f2*lm0[2] + f3*lm0[3];
        float m1 = f0*lm1[0] + f1*lm1[1] + f2*lm1[2] + f3*lm1[3];
        float o0 = a0 * (1.f + m0);
        float o1 = a1 * (1.f + m1);
        float s = o0*o0 + o1*o1;
        for (int m = 16; m > 0; m >>= 1) s += __shfl_xor(s, m, 32);
        float nrm = 1.f / fmaxf(sqrtf(s), 1e-12f);
        o0 *= nrm; o1 *= nrm;
        u_cur[base+c]   = o0;
        u_cur[base+c+1] = o1;
        u_res[base+c]   += o0;
        u_res[base+c+1] += o1;
    }
}

// ---------------------------------------------------------------------------
// Distance-correlation over NF=4 latent rows (6 pairs of 64x64), 1 block.
// add_norm: input rows l -> use normalize(l)+l (stack 1 semantics).
// ---------------------------------------------------------------------------
__global__ void cor_kernel(const float* __restrict__ lat, int add_norm,
                           float* __restrict__ out)
{
    __shared__ float L[NFAC][CDIM];
    __shared__ float am[64*64];
    __shared__ float bm[64*64];
    __shared__ float rma[64], rmb[64];
    __shared__ float tot[2];
    __shared__ float s_ab[8], s_aa[8], s_bb[8];

    const int t = threadIdx.x;
    if (t < NFAC*CDIM) L[t>>6][t&63] = lat[t];
    __syncthreads();
    if (add_norm && t < NFAC) {
        float s = 0.f;
        for (int k = 0; k < CDIM; ++k) { float v = L[t][k]; s += v*v; }
        float inv = 1.f / fmaxf(sqrtf(s), 1e-12f);
        for (int k = 0; k < CDIM; ++k) L[t][k] = L[t][k]*inv + L[t][k];
    }
    __syncthreads();

    const int pi[6] = {0,0,0,1,1,2};
    const int pj[6] = {1,2,3,2,3,3};
    float corsum = 0.f;
    for (int pr = 0; pr < 6; ++pr) {
        int i = pi[pr], j = pj[pr];
        __syncthreads();
        for (int o = t; o < 4096; o += 256) {
            int p = o >> 6, q = o & 63;
            float d1 = L[i][p] - L[i][q];
            float d2 = L[j][p] - L[j][q];
            am[o] = sqrtf(d1*d1 + 1e-8f);
            bm[o] = sqrtf(d2*d2 + 1e-8f);
        }
        __syncthreads();
        if (t < 64) {
            float sa = 0.f, sb = 0.f;
            for (int q = 0; q < 64; ++q) { sa += am[t*64+q]; sb += bm[t*64+q]; }
            rma[t] = sa * (1.f/64.f);
            rmb[t] = sb * (1.f/64.f);
        }
        __syncthreads();
        if (t == 0) {
            float sa = 0.f, sb = 0.f;
            for (int q = 0; q < 64; ++q) { sa += rma[q]; sb += rmb[q]; }
            tot[0] = sa * (1.f/64.f);
            tot[1] = sb * (1.f/64.f);
        }
        __syncthreads();
        float tma = tot[0], tmb = tot[1];
        float pab = 0.f, paa = 0.f, pbb = 0.f;
        for (int o = t; o < 4096; o += 256) {
            int p = o >> 6, q = o & 63;
            float A = am[o] - rma[p] - rma[q] + tma;
            float B = bm[o] - rmb[p] - rmb[q] + tmb;
            pab += A*B; paa += A*A; pbb += B*B;
        }
        for (int m = 16; m > 0; m >>= 1) {
            pab += __shfl_xor(pab, m, 32);
            paa += __shfl_xor(paa, m, 32);
            pbb += __shfl_xor(pbb, m, 32);
        }
        if ((t & 31) == 0) { s_ab[t>>5] = pab; s_aa[t>>5] = paa; s_bb[t>>5] = pbb; }
        __syncthreads();
        if (t == 0) {
            float ab = 0.f, aa = 0.f, bb = 0.f;
            for (int w = 0; w < 8; ++w) { ab += s_ab[w]; aa += s_aa[w]; bb += s_bb[w]; }
            float dab = sqrtf(fmaxf(ab*(1.f/4096.f), 0.f) + 1e-8f);
            float daa = sqrtf(fmaxf(aa*(1.f/4096.f), 0.f) + 1e-8f);
            float dbb = sqrtf(fmaxf(bb*(1.f/4096.f), 0.f) + 1e-8f);
            corsum += dab / sqrtf(daa*dbb + 1e-8f);
        }
        __syncthreads();
    }
    if (t == 0) *out = corsum;
}

// ---------------------------------------------------------------------------
extern "C" void kernel_launch(void* const* d_in, const int* in_sizes, int n_in,
                              void* d_out, int out_size, void* d_ws, size_t ws_size,
                              hipStream_t stream)
{
    const float* user_emb   = (const float*)d_in[0];
    const float* entity_emb = (const float*)d_in[1];
    const float* latent_emb = (const float*)d_in[2];
    const float* latent_div = (const float*)d_in[3];
    const float* weight     = (const float*)d_in[4];
    const float* weight_d   = (const float*)d_in[5];
    const float* cW1  = (const float*)d_in[6];
    const float* cb1  = (const float*)d_in[7];
    const float* cW2  = (const float*)d_in[8];
    const float* cb2  = (const float*)d_in[9];
    const float* cWua = (const float*)d_in[10];
    const float* cbua = (const float*)d_in[11];
    const float* cWwa = (const float*)d_in[12];
    const float* cbwa = (const float*)d_in[13];
    const float* eW1  = (const float*)d_in[14];
    const float* eb1  = (const float*)d_in[15];
    const float* eW2  = (const float*)d_in[16];
    const float* eb2  = (const float*)d_in[17];
    const float* eWua = (const float*)d_in[18];
    const float* ebua = (const float*)d_in[19];
    const float* eWwa = (const float*)d_in[20];
    const float* ebwa = (const float*)d_in[21];
    const float* vals = (const float*)d_in[22];
    const int* edge_index = (const int*)d_in[23];
    const int* edge_type  = (const int*)d_in[24];
    const int* cate       = (const int*)d_in[25];
    const int* irows      = (const int*)d_in[26];
    const int* icols      = (const int*)d_in[27];

    const int n_usr = in_sizes[0] / CDIM;   // 50000
    const int n_ent = in_sizes[1] / CDIM;   // 100000
    const int R1    = in_sizes[4] / CDIM;   // 15
    const int R2    = in_sizes[5] / CDIM;   // 47
    const int nnz   = in_sizes[22];         // 1e6
    const int nE    = in_sizes[23] / 2;     // 1.5e6

    const int* head = edge_index;
    const int* tail = edge_index + nE;

    float* out = (float*)d_out;
    const size_t esz = (size_t)n_ent * CDIM;
    const size_t usz = (size_t)n_usr * CDIM;
    float* e_res_c = out;
    float* u_res_c = out + esz;
    float* cor_c   = out + esz + usz;
    float* e_res_d = out + esz + usz + 1;
    float* u_res_d = out + 2*esz + usz + 1;
    float* cor_d   = out + 2*esz + 2*usz + 1;

    // workspace layout
    float* ws = (float*)d_ws;
    size_t off = 0;
    float* e_cur   = ws + off; off += esz;
    float* u_cur   = ws + off; off += usz;
    float* ssum    = ws + off; off += esz;
    float* cntb    = ws + off; off += (size_t)n_ent;
    float* u_agg   = ws + off; off += usz;
    float* lat_cur = ws + off; off += 256;
    float* lat_mix = ws + off; off += 256;
    float* lat_acc = ws + off; off += 256;
    float* cvec    = ws + off; off += 16;
    _Float16* Mt   = (_Float16*)(ws + off); off += 1024;
    (void)ws_size; (void)n_in; (void)out_size;

    const int NH = 2;
    for (int s = 0; s < 2; ++s) {
        const float* W1  = s ? eW1  : cW1;
        const float* b1  = s ? eb1  : cb1;
        const float* W2  = s ? eW2  : cW2;
        const float* b2  = s ? eb2  : cb2;
        const float* Wua = s ? eWua : cWua;
        const float* bua = s ? ebua : cbua;
        const float* Wwa = s ? eWwa : cWwa;
        const float* bwa = s ? ebwa : cbwa;
        const float* wsel = s ? weight_d : weight;
        const int R = s ? R2 : R1;
        const float* l0 = s ? latent_div : latent_emb;
        float* e_res = s ? e_res_d : e_res_c;
        float* u_res = s ? u_res_d : u_res_c;
        float* cor_o = s ? cor_d   : cor_c;

        hipMemcpyAsync(e_cur, entity_emb, esz*sizeof(float), hipMemcpyDeviceToDevice, stream);
        hipMemcpyAsync(u_cur, user_emb,   usz*sizeof(float), hipMemcpyDeviceToDevice, stream);
        hipMemcpyAsync(lat_cur, l0, NFAC*CDIM*sizeof(float), hipMemcpyDeviceToDevice, stream);
        hipMemcpyAsync(e_res, entity_emb, esz*sizeof(float), hipMemcpyDeviceToDevice, stream);
        hipMemcpyAsync(u_res, user_emb,   usz*sizeof(float), hipMemcpyDeviceToDevice, stream);
        if (s) hipMemcpyAsync(lat_acc, latent_div, NFAC*CDIM*sizeof(float),
                              hipMemcpyDeviceToDevice, stream);

        for (int i = 0; i < NH; ++i) {
            hipMemsetAsync(ssum, 0, esz*sizeof(float), stream);
            hipMemsetAsync(cntb, 0, (size_t)n_ent*sizeof(float), stream);
            hipMemsetAsync(u_agg, 0, usz*sizeof(float), stream);

            setup_hop<<<1, 256, 0, stream>>>(
                lat_cur, W1 + i*CDIM*CDIM, b1 + i*CDIM,
                W2 + i*CDIM*CDIM, b2 + i*CDIM,
                Wwa + i*R*R, bwa + i*R, wsel, R,
                Mt, cvec, lat_mix, lat_cur, s ? lat_acc : nullptr, s);

            {
                long long th = (long long)nE * 32;
                int blocks = (int)((th + 255) / 256);
                edge_agg<<<blocks, 256, 0, stream>>>(
                    e_cur, wsel, head, tail, edge_type, cate, ssum, cntb, nE, s);
            }
            {
                long long th = (long long)nnz * 32;
                int blocks = (int)((th + 255) / 256);
                inter_agg<<<blocks, 256, 0, stream>>>(
                    e_cur, vals, irows, icols, u_agg, nnz);
            }
            user_finalize<<<(n_usr + 15) / 16, 32, 0, stream>>>(
                u_cur, u_agg, Mt, cvec, Wua + i*NFAC*NFAC, bua + i*NFAC,
                lat_mix, u_res, n_usr);
            {
                long long th = (long long)n_ent * 32;
                int blocks = (int)((th + 255) / 256);
                ent_finalize<<<blocks, 256, 0, stream>>>(
                    ssum, cntb, e_cur, e_res, n_ent);
            }
        }
        // stack1: cor(normalize(l)+l); stack2: cor(lat_acc)
        cor_kernel<<<1, 256, 0, stream>>>(s ? lat_acc : lat_cur, s ? 0 : 1, cor_o);
    }
}
